// LambdaLayer_75325136437778
// MI455X (gfx1250) — matrix operand order repaired and checked by reading, not compiled, not run on hardware
//
#include <hip/hip_runtime.h>
#include <cstdint>

typedef _Float16 v16h __attribute__((ext_vector_type(16)));
typedef float    v8f  __attribute__((ext_vector_type(8)));

#define B_     32
#define DIM_   512
#define N_     1024      // H*W
#define DK_    16
#define HEADS_ 4
#define DV_    128
#define R_     15
#define TAPS_  225       // 15*15, padded to 256 for K
#define ROWS_  208       // 64 q + 16 k + 128 v

// ---------------- workspace layout (byte offsets, 256-aligned) ----------------
#define OFF_SCALE  0u            //  208 f32
#define OFF_BIAS   1024u         //  208 f32
#define OFF_EMBA   2048u         //  8*32*16 f16      (8192 B)
#define OFF_WALLA  10240u        //  13*16*32*16 f16  (212992 B)
#define OFF_XB     223232u       //  32*64*16*32*16 f16 (33554432 B)
#define OFF_Q      33777664u     //  32*64*1024 f32   (8388608 B)
#define OFF_K      42166272u     //  32*16*1024 f32   (2097152 B)
#define OFF_V      44263424u     //  32*128*1024 f32  (16777216 B)
#define OFF_CL     61040640u     //  32*16*128 f32    (262144 B)
#define OFF_VH     61302784u     //  32*128*1024 f16  (8388608 B)
// total ~69.7 MB

__device__ __forceinline__ int koffA(int lane, int j) {
    // 16-bit A-matrix 16x32 lane layout (contiguous K order):
    // lane<16: halfs 0..15 -> K=0..15; lane>=16: K=16..31
    return ((lane >= 16) ? 16 : 0) + j;
}

// ---------- kernel 0a: fold BN into per-row scale/bias of fused W ----------
__global__ void k_scalebias(const float* qg, const float* qb,
                            const float* vg, const float* vb,
                            float* scale, float* bias) {
    int r = blockIdx.x * blockDim.x + threadIdx.x;
    if (r >= ROWS_) return;
    float inv = rsqrtf(1.0f + 1e-5f);
    float s, b;
    if (r < 64)      { s = qg[r] * inv;      b = qb[r]; }
    else if (r < 80) { s = 1.0f;             b = 0.0f; }
    else             { s = vg[r - 80] * inv; b = vb[r - 80]; }
    scale[r] = s; bias[r] = b;
}

// ---------- kernel 0b: swizzle fused weight (208x512) into A-fragments ----------
__global__ void k_wallA(const float* wq, const float* wk, const float* wv,
                        _Float16* wallA) {
    int tid  = blockIdx.x * blockDim.x + threadIdx.x;   // 26*256 = 6656 exact
    int lane = tid & 31;
    int s    = (tid >> 5) & 15;                         // k-step (K=32 each)
    int mt   = tid >> 9;                                // M tile 0..12
    int row  = mt * 16 + (lane & 15);
    _Float16 frag[16];
#pragma unroll
    for (int j = 0; j < 16; ++j) {
        int K = s * 32 + koffA(lane, j);
        float w;
        if (row < 64)      w = wq[row * DIM_ + K];
        else if (row < 80) w = wk[(row - 64) * DIM_ + K];
        else               w = wv[(row - 80) * DIM_ + K];
        frag[j] = (_Float16)w;
    }
    _Float16* dst = wallA + ((size_t)tid << 4);
#pragma unroll
    for (int j = 0; j < 16; ++j) dst[j] = frag[j];
}

// ---------- kernel 0c: swizzle position embedding (16x225 -> A-frags, K padded 256) ----------
__global__ void k_embA(const float* emb, _Float16* embA) {
    int tid  = threadIdx.x;                             // 256 threads exact
    int lane = tid & 31;
    int s    = tid >> 5;                                // 0..7
    int row  = lane & 15;                               // k-dim 0..15
    _Float16* dst = embA + ((size_t)tid << 4);
#pragma unroll
    for (int j = 0; j < 16; ++j) {
        int t = s * 32 + koffA(lane, j);
        dst[j] = (t < TAPS_) ? (_Float16)emb[row * TAPS_ + t] : (_Float16)0.0f;
    }
}

// ---------- kernel 0d: transpose-gather x into B-fragments ----------
// B-fragment (32x16, f16): lane n (0..15) = column n, 16 halfs = K ascending;
// lane n+16 = column n, K+16..K+31.
__global__ void k_xB(const float* x, _Float16* xB) {
    int tid   = blockIdx.x * blockDim.x + threadIdx.x;  // 4096*256 = 1,048,576 exact
    int n     = tid & 1023;
    int chunk = (tid >> 10) & 31;                       // d-chunk of 16
    int b     = tid >> 15;
    int d0    = chunk * 16;
    int s     = chunk >> 1;
    int hi    = chunk & 1;
    int l     = (n & 15) + (hi ? 16 : 0);
    int t     = n >> 4;
    const float* src = x + ((size_t)b * DIM_) * N_ + n;
    _Float16 frag[16];
#pragma unroll
    for (int j = 0; j < 16; ++j)
        frag[j] = (_Float16)src[(size_t)(d0 + j) * N_];
    _Float16* dst = xB + ((((((size_t)b * 64 + t) * 16 + s) * 32 + l)) << 4);
#pragma unroll
    for (int j = 0; j < 16; ++j) dst[j] = frag[j];
}

// ---------- kernel 1: fused projection GEMM  (208 x 512) x (512 x 1024) per batch ----------
__global__ __launch_bounds__(256) void k_gemm(const _Float16* __restrict__ wallA,
                                              const _Float16* __restrict__ xB,
                                              const float* __restrict__ scale,
                                              const float* __restrict__ bias,
                                              float* qbuf, float* kbuf, float* vbuf,
                                              _Float16* vhbuf) {
    int wave  = threadIdx.x >> 5;
    int lane  = threadIdx.x & 31;
    int ntile = blockIdx.x * 8 + wave;                  // 0..63
    int mtile = blockIdx.y;                             // 0..12
    int b     = blockIdx.z;                             // 0..31

    const _Float16* bBase = xB + ((((size_t)b * 64 + ntile) * 16) << 9); // *32*16
    v8f acc = {};
#pragma unroll 4
    for (int s = 0; s < 16; ++s) {
        v16h a  = *(const v16h*)(wallA + ((((size_t)(mtile * 16 + s)) * 32 + lane) << 4));
        v16h bb = *(const v16h*)(bBase + ((((size_t)s) * 32 + lane) << 4));
        if (s + 1 < 16)
            __builtin_prefetch(bBase + ((((size_t)(s + 1)) * 32 + lane) << 4), 0, 3);
        acc = __builtin_amdgcn_wmma_f32_16x16x32_f16(false, a, false, bb,
                                                     (short)0, acc, false, false);
    }
    int hi  = lane >> 4;
    int col = lane & 15;
    int n   = ntile * 16 + col;
#pragma unroll
    for (int r = 0; r < 8; ++r) {
        int row = mtile * 16 + r + hi * 8;
        float v = acc[r] * scale[row] + bias[row];
        if (row < 64)      qbuf[((size_t)b * 64 + row) * N_ + n] = v;
        else if (row < 80) kbuf[((size_t)b * 16 + (row - 64)) * N_ + n] = v;
        else {
            vbuf [((size_t)b * DV_ + (row - 80)) * N_ + n] = v;
            vhbuf[((size_t)b * DV_ + (row - 80)) * N_ + n] = (_Float16)v;
        }
    }
}

// ---------- kernel 2: softmax over k-channels + content lambda ----------
__global__ __launch_bounds__(256) void k_cl(const float* __restrict__ kbuf,
                                            const float* __restrict__ vbuf,
                                            float* clbuf) {
    int b   = blockIdx.x;
    int tid = threadIdx.x;
    __shared__ _Float16 kf[DK_ * N_];                   // 32 KB
    for (int i = 0; i < 4; ++i) {
        int n = tid + i * 256;
        float vals[DK_]; float m = -1e30f;
#pragma unroll
        for (int kk = 0; kk < DK_; ++kk) {
            vals[kk] = kbuf[((size_t)b * DK_ + kk) * N_ + n];
            m = fmaxf(m, vals[kk]);
        }
        float ssum = 0.0f;
#pragma unroll
        for (int kk = 0; kk < DK_; ++kk) { vals[kk] = __expf(vals[kk] - m); ssum += vals[kk]; }
        float inv = 1.0f / ssum;
#pragma unroll
        for (int kk = 0; kk < DK_; ++kk) kf[kk * N_ + n] = (_Float16)(vals[kk] * inv);
    }
    __syncthreads();
    int w = tid >> 5, lane = tid & 31;
    for (int i = 0; i < 16; ++i) {
        int v = w * 16 + i;
        const float* vp = vbuf + ((size_t)b * DV_ + v) * N_;
        float acc[DK_] = {};
        for (int n0 = 0; n0 < N_; n0 += 32) {
            float vv = vp[n0 + lane];
#pragma unroll
            for (int kk = 0; kk < DK_; ++kk)
                acc[kk] += (float)kf[kk * N_ + n0 + lane] * vv;
        }
#pragma unroll
        for (int kk = 0; kk < DK_; ++kk) {
            float a = acc[kk];
            for (int off = 16; off > 0; off >>= 1) a += __shfl_xor(a, off);
            if (lane == 0) clbuf[((size_t)b * DK_ + kk) * DV_ + v] = a;
        }
    }
}

// ---------- kernel 3: position-lambda conv (WMMA) fused with output contraction ----------
// block = (v-chunk of 32, image row y, batch b); 256 threads = 8 waves
__global__ __launch_bounds__(256) void k_out(const float* __restrict__ qbuf,
                                             const _Float16* __restrict__ vhbuf,
                                             const float* __restrict__ clbuf,
                                             const _Float16* __restrict__ embA,
                                             float* __restrict__ out) {
    int chunk = blockIdx.x;            // 0..3  -> ch0 = chunk*32
    int y0    = blockIdx.y;            // 0..31
    int b     = blockIdx.z;            // 0..31
    int ch0   = chunk * 32;
    int tid   = threadIdx.x;
    int wave  = tid >> 5, lane = tid & 31;
    int hi    = lane >> 4, col = lane & 15;

    // halo column c holds v at gx = c - 8 (valid span c in [8,40) -> 16B-aligned)
    __shared__ __align__(32) _Float16 vhalo[32 * 15 * 48];   // 46080 B
    __shared__ __align__(32) _Float16 embA_lds[8 * 32 * 16]; //  8192 B
    __shared__ float    q_lds[64 * 32];                      //  8192 B
    __shared__ float    cl_lds[DK_ * 32];                    //  2048 B
    __shared__ unsigned short tapOff[256];                   //   512 B

    // ---- phase 1: zero the halo (covers padding + out-of-image rows) ----
    {
        uint4* z = (uint4*)vhalo;                            // 2880 x 16B
        uint4 zero = {0u, 0u, 0u, 0u};
        for (int idx = tid; idx < 2880; idx += 256) z[idx] = zero;
    }
    __syncthreads();

    // ---- phase 2: async-copy valid halo interiors (global f16 -> LDS), plus
    //      regular staging of q tile, cl tile, emb A-fragments, tap offsets ----
    for (int idx = tid; idx < 32 * 15 * 4; idx += 256) {     // (ch, dy, 16B segment)
        int ch  = idx / 60;
        int rem = idx - ch * 60;
        int dy  = rem >> 2;
        int seg = rem & 3;
        int gy  = y0 - 7 + dy;
        if ((unsigned)gy < 32u) {
            const _Float16* gsrc = vhbuf + ((size_t)b * DV_ + ch0 + ch) * N_
                                         + gy * 32 + seg * 8;
            uint32_t lds_off = (uint32_t)(uintptr_t)(vhalo + ch * 720 + dy * 48 + 8 + seg * 8);
            uint64_t gaddr   = (uint64_t)(uintptr_t)gsrc;
            asm volatile("global_load_async_to_lds_b128 %0, %1, off"
                         :: "v"(lds_off), "v"(gaddr) : "memory");
        }
    }
    for (int idx = tid; idx < 64 * 32; idx += 256) {
        int row = idx >> 5, xx = idx & 31;
        q_lds[idx] = qbuf[((size_t)b * 64 + row) * N_ + y0 * 32 + xx];
    }
    for (int idx = tid; idx < DK_ * 32; idx += 256) {
        int kk = idx >> 5, ch = idx & 31;
        cl_lds[idx] = clbuf[((size_t)b * DK_ + kk) * DV_ + ch0 + ch];
    }
    for (int idx = tid; idx < 8 * 32; idx += 256) {
        const _Float16* src = embA + ((size_t)idx << 4);
        _Float16*       dst = embA_lds + (idx << 4);
#pragma unroll
        for (int j = 0; j < 16; ++j) dst[j] = src[j];
    }
    {
        int t = tid;
        tapOff[t] = (t < TAPS_) ? (unsigned short)((t / R_) * 48 + (t % R_) + 1)
                                : (unsigned short)0xFFFFu;
    }
    asm volatile("s_wait_asynccnt 0x0" ::: "memory");
    __syncthreads();

    // ---- each wave: 8 tasks of (channel, column-tile) ----
    for (int it = 0; it < 8; ++it) {
        int task    = wave * 8 + it;       // 0..63
        int ch      = task >> 1;           // 0..31
        int coltile = task & 1;            // 0..1
        int x       = coltile * 16 + col;  // lane's output column 0..31
        int chBase  = ch * 720;

        v8f acc = {};
        int tb = (hi ? 16 : 0);
#pragma unroll
        for (int s = 0; s < 8; ++s) {
            v16h a = *(const v16h*)(embA_lds + ((s * 32 + lane) << 4));
            v16h bb;
#pragma unroll
            for (int j = 0; j < 16; ++j) {
                int t = s * 32 + tb + j;
                _Float16 v = (_Float16)0.0f;
                if (t < TAPS_) {
                    unsigned off = tapOff[t];
                    v = vhalo[chBase + (int)off + x];
                }
                bb[j] = v;
            }
            acc = __builtin_amdgcn_wmma_f32_16x16x32_f16(false, a, false, bb,
                                                         (short)0, acc, false, false);
        }

        // ---- epilogue: out[h,v,n] = sum_k q[h*16+k,n] * (cl[k,v] + pl[k,v,n]) ----
        float o0 = 0.f, o1 = 0.f, o2 = 0.f, o3 = 0.f;
        int kb = hi * 8;
#pragma unroll
        for (int r = 0; r < 8; ++r) {
            int kk  = kb + r;
            float t = acc[r] + cl_lds[kk * 32 + ch];
            o0 += q_lds[(0 * 16 + kk) * 32 + x] * t;
            o1 += q_lds[(1 * 16 + kk) * 32 + x] * t;
            o2 += q_lds[(2 * 16 + kk) * 32 + x] * t;
            o3 += q_lds[(3 * 16 + kk) * 32 + x] * t;
        }
        o0 += __shfl_xor(o0, 16);
        o1 += __shfl_xor(o1, 16);
        o2 += __shfl_xor(o2, 16);
        o3 += __shfl_xor(o3, 16);
        if (!hi) {
            size_t base = ((size_t)b * (HEADS_ * DV_)) * N_ + (size_t)y0 * 32 + x;
            int c = ch0 + ch;
            out[base + (size_t)(0 * DV_ + c) * N_] = o0;
            out[base + (size_t)(1 * DV_ + c) * N_] = o1;
            out[base + (size_t)(2 * DV_ + c) * N_] = o2;
            out[base + (size_t)(3 * DV_ + c) * N_] = o3;
        }
    }
}

// -------------------------------- launcher --------------------------------
extern "C" void kernel_launch(void* const* d_in, const int* in_sizes, int n_in,
                              void* d_out, int out_size, void* d_ws, size_t ws_size,
                              hipStream_t stream) {
    const float* x   = (const float*)d_in[0];
    const float* wq  = (const float*)d_in[1];
    const float* wk  = (const float*)d_in[2];
    const float* wv  = (const float*)d_in[3];
    const float* emb = (const float*)d_in[4];
    const float* qg  = (const float*)d_in[5];
    const float* qb  = (const float*)d_in[6];
    const float* vg  = (const float*)d_in[7];
    const float* vb  = (const float*)d_in[8];
    float* out = (float*)d_out;

    char* ws = (char*)d_ws;
    float*    scaleB = (float*)(ws + OFF_SCALE);
    float*    biasB  = (float*)(ws + OFF_BIAS);
    _Float16* embA   = (_Float16*)(ws + OFF_EMBA);
    _Float16* wallA  = (_Float16*)(ws + OFF_WALLA);
    _Float16* xB     = (_Float16*)(ws + OFF_XB);
    float*    qbuf   = (float*)(ws + OFF_Q);
    float*    kbuf   = (float*)(ws + OFF_K);
    float*    vbuf   = (float*)(ws + OFF_V);
    float*    clbuf  = (float*)(ws + OFF_CL);
    _Float16* vhbuf  = (_Float16*)(ws + OFF_VH);

    k_scalebias<<<1, 256, 0, stream>>>(qg, qb, vg, vb, scaleB, biasB);
    k_wallA<<<26, 256, 0, stream>>>(wq, wk, wv, wallA);
    k_embA<<<1, 256, 0, stream>>>(emb, embA);
    k_xB<<<4096, 256, 0, stream>>>(x, xB);
    k_gemm<<<dim3(8, 13, 32), 256, 0, stream>>>(wallA, xB, scaleB, biasB,
                                                qbuf, kbuf, vbuf, vhbuf);
    k_cl<<<32, 256, 0, stream>>>(kbuf, vbuf, clbuf);
    k_out<<<dim3(4, 32, 32), 256, 0, stream>>>(qbuf, vhbuf, clbuf, embA, out);
}
